// MultiModalMoE_44684839748103
// MI455X (gfx1250) — compile-verified, matrix operand order checked by reading, not run on hardware
//
#include <hip/hip_runtime.h>
#include <hip/hip_bf16.h>

// ---------------------------------------------------------------------------
// MultiModal MoE block for MI455X (gfx1250, wave32, WMMA).
// All GEMMs run through v_wmma_f32_16x16x32_bf16 (f32 accumulate).
// GEMM: 128x128x32 block tiles, 8 waves (2x4), 8 WMMAs/wave/k-step,
// double-buffered LDS (one barrier per k-step) with a 2-slab register
// prefetch pipeline, branch-free b128 staging.
// Requires M%128==0, N%128==0, K%32==0 (true for every launch here).
// Workspace requirement: ~202 MB (see offsets in kernel_launch).
// ---------------------------------------------------------------------------

#define B_    8
#define S_    1024
#define E_    768
#define H_    12
#define NE_   8
#define F_    (4 * E_)     // 3072
#define DH_   (E_ / H_)    // 64
#define NTOK  (B_ * S_)    // 8192

typedef __bf16 bf16_t;
typedef __attribute__((ext_vector_type(4)))  __bf16 v4bf;
typedef __attribute__((ext_vector_type(16))) __bf16 v16bf;
typedef __attribute__((ext_vector_type(8)))  float  v8f;

__device__ __forceinline__ float gelu_exact(float x) {
    return 0.5f * x * (1.0f + erff(x * 0.70710678118654752440f));
}

// Offset of element (ml, k) inside a 16x32 (MxK) bf16 WMMA operand tile stored
// in "lane-major" LDS layout: 32 lanes x 16 bf16 each (1024 B per tile).
// Per ISA 7.12.2 (16-bit A-Matrix 16x32):
//   lanes 0-15 : M=ml, VGPRs 0-3 hold K=0..7 (pairs), VGPRs 4-7 hold K=16..23
//   lanes 16-31: M=ml, VGPRs 0-3 hold K=8..15,        VGPRs 4-7 hold K=24..31
// A 4-aligned k-chunk maps to 4 consecutive bf16 slots within one lane.
__device__ __forceinline__ int frag_off(int ml, int k) {
    int hi   = (k >= 16) ? 4 : 0;
    int kk   = k & 15;
    int lane = ml + ((kk >> 3) << 4);
    int v    = hi + ((kk >> 1) & 3);
    int idx  = (v << 1) | (kk & 1);
    return lane * 16 + idx;
}

__device__ __forceinline__ v4bf pack4(float4 f) {
    v4bf r;
    r[0] = (bf16_t)f.x; r[1] = (bf16_t)f.y; r[2] = (bf16_t)f.z; r[3] = (bf16_t)f.w;
    return r;
}

// ---------------------------------------------------------------------------
// LayerNorm: one block per token, E=768.
// ---------------------------------------------------------------------------
__global__ __launch_bounds__(256) void ln_kernel(const float* __restrict__ x,
                                                 const float* __restrict__ g,
                                                 const float* __restrict__ b,
                                                 float* __restrict__ out) {
    const int tok = blockIdx.x;
    const int tid = threadIdx.x;
    const float* xr = x + (size_t)tok * E_;
    float* orow = out + (size_t)tok * E_;
    __shared__ float red[256];

    float s = 0.f;
    for (int i = tid; i < E_; i += 256) s += xr[i];
    red[tid] = s; __syncthreads();
    for (int st = 128; st > 0; st >>= 1) { if (tid < st) red[tid] += red[tid + st]; __syncthreads(); }
    float mean = red[0] * (1.0f / E_);
    __syncthreads();

    float v = 0.f;
    for (int i = tid; i < E_; i += 256) { float d = xr[i] - mean; v += d * d; }
    red[tid] = v; __syncthreads();
    for (int st = 128; st > 0; st >>= 1) { if (tid < st) red[tid] += red[tid + st]; __syncthreads(); }
    float inv = rsqrtf(red[0] * (1.0f / E_) + 1e-5f);

    for (int i = tid; i < E_; i += 256)
        orow[i] = (xr[i] - mean) * inv * g[i] + b[i];
}

// ---------------------------------------------------------------------------
// WMMA GEMM: C[M,N] = epilogue( A[M,K] @ W[N,K]^T + bias )
// Block 256 threads = 8 waves (2 Mw x 4 Nw); block tile 128x128, k-step 32.
// Wave tile 64x32 = 4x2 WMMA tiles. Double-buffered LDS: while WMMAs read
// buffer `cur`, the slab prefetched one iteration ago is converted/stored
// into buffer `cur^1`, and the slab for iteration s+2 is loaded from global.
// Exactly one __syncthreads() per k-step covers both hazards.
// Epilogue: +bias -> optional exact GELU -> *scale[m] -> +residual ->
// write or accumulate.
// ---------------------------------------------------------------------------
__global__ __launch_bounds__(256) void wmma_gemm_kernel(
        const float* __restrict__ A, const float* __restrict__ W,
        const float* __restrict__ bias, const float* __restrict__ residual,
        const float* __restrict__ scale, float* __restrict__ out,
        int M, int N, int K, int do_gelu, int accumulate) {
    __shared__ alignas(32) bf16_t smA[2][8 * 512];   // 2 x 8 KB
    __shared__ alignas(32) bf16_t smB[2][8 * 512];   // 2 x 8 KB

    const int tid  = threadIdx.x;
    const int lane = tid & 31;
    const int w    = tid >> 5;
    const int mw   = w & 1;       // 0..1 : 64-row half
    const int nw   = w >> 1;      // 0..3 : 32-col strip
    const int m0   = blockIdx.y * 128;
    const int n0   = blockIdx.x * 128;

    // Per-thread staging slots: 4 chunks of A, 4 of W. Chunk c handles
    // element i = tid + 256*c of 1024 chunks; row = i>>3, kc = (i&7)*4.
    const float* agp[4]; const float* bgp[4];
    int lds_off[4];
    #pragma unroll
    for (int c = 0; c < 4; ++c) {
        int i  = tid + 256 * c;
        int r  = i >> 3;
        int kc = (i & 7) << 2;
        agp[c]    = A + (size_t)(m0 + r) * K + kc;
        bgp[c]    = W + (size_t)(n0 + r) * K + kc;
        lds_off[c] = ((r >> 4) << 9) + frag_off(r & 15, kc);
    }

    v8f acc[4][2];
    #pragma unroll
    for (int i = 0; i < 4; ++i)
        #pragma unroll
        for (int j = 0; j < 2; ++j) acc[i][j] = (v8f){};

    const int nslab = K >> 5;
    float4 ra[4], rb[4];

    // Prologue: slab 0 -> regs -> LDS buf 0; slab 1 -> regs.
    #pragma unroll
    for (int c = 0; c < 4; ++c) { ra[c] = *(const float4*)agp[c]; rb[c] = *(const float4*)bgp[c]; }
    #pragma unroll
    for (int c = 0; c < 4; ++c) {
        *(v4bf*)&smA[0][lds_off[c]] = pack4(ra[c]);
        *(v4bf*)&smB[0][lds_off[c]] = pack4(rb[c]);
    }
    if (nslab > 1) {
        #pragma unroll
        for (int c = 0; c < 4; ++c) {
            ra[c] = *(const float4*)(agp[c] + 32);
            rb[c] = *(const float4*)(bgp[c] + 32);
        }
    }
    __syncthreads();

    for (int s = 0; s < nslab; ++s) {
        const int cur = s & 1;

        // Pipeline: commit slab s+1 into the other buffer; fetch slab s+2.
        if (s + 1 < nslab) {
            #pragma unroll
            for (int c = 0; c < 4; ++c) {
                *(v4bf*)&smA[cur ^ 1][lds_off[c]] = pack4(ra[c]);
                *(v4bf*)&smB[cur ^ 1][lds_off[c]] = pack4(rb[c]);
            }
            if (s + 2 < nslab) {
                #pragma unroll
                for (int c = 0; c < 4; ++c) {
                    ra[c] = *(const float4*)(agp[c] + (s + 2) * 32);
                    rb[c] = *(const float4*)(bgp[c] + (s + 2) * 32);
                }
            }
        }

        // Fragment loads + 8 WMMAs (4 A-tiles x 2 B-tiles) from buffer `cur`.
        v16bf af[4], bf[2];
        #pragma unroll
        for (int i = 0; i < 4; ++i)
            af[i] = *reinterpret_cast<const v16bf*>(&smA[cur][((mw * 4 + i) << 9) + lane * 16]);
        #pragma unroll
        for (int j = 0; j < 2; ++j)
            bf[j] = *reinterpret_cast<const v16bf*>(&smB[cur][((nw * 2 + j) << 9) + lane * 16]);

        #pragma unroll
        for (int i = 0; i < 4; ++i)
            #pragma unroll
            for (int j = 0; j < 2; ++j)
                acc[i][j] = __builtin_amdgcn_wmma_f32_16x16x32_bf16(
                    false, af[i], false, bf[j], (short)0, acc[i][j], false, false);

        __syncthreads();
    }

    // C/D layout: lane&15 = N, VGPR r -> M = r + 8*(lane>>4).
    const int nloc  = lane & 15;
    const int mbase = (lane >> 4) << 3;
    #pragma unroll
    for (int i = 0; i < 4; ++i) {
        #pragma unroll
        for (int r = 0; r < 8; ++r) {
            int m = m0 + mw * 64 + i * 16 + mbase + r;
            float sc = scale ? scale[m] : 1.0f;
            #pragma unroll
            for (int j = 0; j < 2; ++j) {
                int n = n0 + nw * 32 + j * 16 + nloc;
                float val = acc[i][j][r];
                if (bias) val += bias[n];
                if (do_gelu) val = gelu_exact(val);
                val *= sc;
                size_t o = (size_t)m * N + n;
                if (residual) val += residual[o];
                if (accumulate) out[o] += val; else out[o] = val;
            }
        }
    }
}

// ---------------------------------------------------------------------------
// Attention: one 64-thread block per (q, b*h). Two-pass softmax over S keys.
// qkv layout [NTOK, 3E]: q at +0, k at +E, v at +2E; head h occupies dims
// [h*DH, h*DH+DH).
// ---------------------------------------------------------------------------
__global__ __launch_bounds__(64) void attention_kernel(const float* __restrict__ qkv,
                                                       float* __restrict__ ctx) {
    const int q   = blockIdx.x;
    const int bh  = blockIdx.y;
    const int b   = bh / H_;
    const int h   = bh % H_;
    const int tid = threadIdx.x;
    const int E3  = 3 * E_;

    __shared__ float qs[DH_];
    __shared__ float sc[S_];
    __shared__ float red[64];

    const float* qrow = qkv + ((size_t)(b * S_ + q)) * E3 + h * DH_;
    if (tid < DH_) qs[tid] = qrow[tid];
    __syncthreads();

    float lmax = -3.4e38f;
    for (int k = tid; k < S_; k += 64) {
        const float* krow = qkv + ((size_t)(b * S_ + k)) * E3 + E_ + h * DH_;
        float d = 0.f;
        #pragma unroll 8
        for (int j = 0; j < DH_; ++j) d += qs[j] * krow[j];
        d *= 0.125f;                       // 1/sqrt(64)
        sc[k] = d;
        lmax = fmaxf(lmax, d);
    }
    red[tid] = lmax; __syncthreads();
    for (int st = 32; st > 0; st >>= 1) { if (tid < st) red[tid] = fmaxf(red[tid], red[tid + st]); __syncthreads(); }
    float mx = red[0];
    __syncthreads();

    float lsum = 0.f;
    for (int k = tid; k < S_; k += 64) {
        float e = __expf(sc[k] - mx);
        sc[k] = e;
        lsum += e;
    }
    red[tid] = lsum; __syncthreads();
    for (int st = 32; st > 0; st >>= 1) { if (tid < st) red[tid] += red[tid + st]; __syncthreads(); }
    float inv = 1.0f / red[0];
    __syncthreads();

    // pass 2: tid = output dim d; V reads coalesced across d.
    float acc = 0.f;
    for (int k = 0; k < S_; ++k)
        acc += sc[k] * qkv[((size_t)(b * S_ + k)) * E3 + 2 * E_ + h * DH_ + tid];
    ctx[((size_t)(b * S_ + q)) * E_ + h * DH_ + tid] = acc * inv;
}

// ---------------------------------------------------------------------------
// Router: logits = xn @ router_w^T + b; softmax(8); top-2; renormalized
// softmax over the top-2 PROBS (matching the reference). Produces per-expert
// combine coefficients coef[e][tok] and modality masks.
// ---------------------------------------------------------------------------
__global__ __launch_bounds__(256) void router_kernel(
        const float* __restrict__ xn, const float* __restrict__ rw,
        const float* __restrict__ rb, const int* __restrict__ modality,
        float* __restrict__ coef, float* __restrict__ vmask,
        float* __restrict__ tmask) {
    int tok = blockIdx.x * blockDim.x + threadIdx.x;
    if (tok >= NTOK) return;

    float lg[NE_];
    #pragma unroll
    for (int e = 0; e < NE_; ++e) lg[e] = rb[e];
    const float* xr = xn + (size_t)tok * E_;
    for (int j = 0; j < E_; ++j) {
        float xv = xr[j];
        #pragma unroll
        for (int e = 0; e < NE_; ++e) lg[e] += xv * rw[e * E_ + j];
    }

    float mx = lg[0];
    #pragma unroll
    for (int e = 1; e < NE_; ++e) mx = fmaxf(mx, lg[e]);
    float p[NE_], sum = 0.f;
    #pragma unroll
    for (int e = 0; e < NE_; ++e) { p[e] = __expf(lg[e] - mx); sum += p[e]; }
    float inv = 1.f / sum;
    #pragma unroll
    for (int e = 0; e < NE_; ++e) p[e] *= inv;

    int i1 = 0;
    #pragma unroll
    for (int e = 1; e < NE_; ++e) if (p[e] > p[i1]) i1 = e;
    int i2 = (i1 == 0) ? 1 : 0;
    #pragma unroll
    for (int e = 0; e < NE_; ++e) if (e != i1 && p[e] > p[i2]) i2 = e;

    float a = p[i1], c = p[i2];
    float m2 = fmaxf(a, c);
    float ea = __expf(a - m2), ec = __expf(c - m2);
    float wden = 1.f / (ea + ec);
    float w1 = ea * wden, w2 = ec * wden;

    #pragma unroll
    for (int e = 0; e < NE_; ++e)
        coef[(size_t)e * NTOK + tok] = (e == i1) ? w1 : ((e == i2) ? w2 : 0.f);

    int mt = modality[tok];
    vmask[tok] = (mt == 0) ? 1.f : 0.f;
    tmask[tok] = (mt == 1) ? 1.f : 0.f;
}

__global__ __launch_bounds__(256) void add_kernel(const float* __restrict__ a,
                                                  const float* __restrict__ b,
                                                  float* __restrict__ out, int n) {
    int i = blockIdx.x * 256 + threadIdx.x;
    if (i < n) out[i] = a[i] + b[i];
}

// ---------------------------------------------------------------------------
static inline void launch_gemm(const float* A, const float* W, const float* bias,
                               const float* residual, const float* scale, float* out,
                               int M, int N, int K, int gelu, int accum,
                               hipStream_t s) {
    dim3 grid((N + 127) / 128, (M + 127) / 128);
    wmma_gemm_kernel<<<grid, 256, 0, s>>>(A, W, bias, residual, scale, out,
                                          M, N, K, gelu, accum);
}

extern "C" void kernel_launch(void* const* d_in, const int* in_sizes, int n_in,
                              void* d_out, int out_size, void* d_ws, size_t ws_size,
                              hipStream_t stream) {
    const float* x          = (const float*)d_in[0];
    const int*   modality   = (const int*)  d_in[1];
    const float* norm1_g    = (const float*)d_in[2];
    const float* norm1_b    = (const float*)d_in[3];
    const float* norm2_g    = (const float*)d_in[4];
    const float* norm2_b    = (const float*)d_in[5];
    const float* in_proj_w  = (const float*)d_in[6];
    const float* in_proj_b  = (const float*)d_in[7];
    const float* out_proj_w = (const float*)d_in[8];
    const float* out_proj_b = (const float*)d_in[9];
    const float* router_w   = (const float*)d_in[10];
    const float* router_b   = (const float*)d_in[11];
    const float* spec_ln_g  = (const float*)d_in[12];
    const float* spec_ln_b  = (const float*)d_in[13];
    const float* spec_w1    = (const float*)d_in[14];
    const float* spec_b1    = (const float*)d_in[15];
    const float* spec_w2    = (const float*)d_in[16];
    const float* spec_b2    = (const float*)d_in[17];
    const float* gen_ln_g   = (const float*)d_in[18];
    const float* gen_ln_b   = (const float*)d_in[19];
    const float* gen_w1     = (const float*)d_in[20];
    const float* gen_b1     = (const float*)d_in[21];
    const float* gen_w2     = (const float*)d_in[22];
    const float* gen_b2     = (const float*)d_in[23];
    float* out = (float*)d_out;

    // Workspace layout (floats). h (NTOK*F) exactly overlaps qkv(3E)+ctx(E).
    float* ws  = (float*)d_ws;
    size_t off = 0;
    float* xn   = ws + off; off += (size_t)NTOK * E_;     // LN1 out, later LN2 out
    float* qkv  = ws + off; off += (size_t)NTOK * F_;     // qkv [NTOK,3E]; reused as h
    float* h    = qkv;
    float* ctx  = qkv + (size_t)NTOK * 3 * E_;
    float* x1   = ws + off; off += (size_t)NTOK * E_;     // post-attention residual
    float* xln  = ws + off; off += (size_t)NTOK * E_;     // per-expert LN out
    float* eo   = ws + off; off += (size_t)NTOK * E_;     // expert_out accumulator
    float* coef = ws + off; off += (size_t)NE_ * NTOK;    // router combine coeffs
    float* vm   = ws + off; off += NTOK;
    float* tm   = ws + off; off += NTOK;
    (void)ws_size; (void)off; (void)in_sizes; (void)n_in; (void)out_size;

    // --- attention block ---
    ln_kernel<<<NTOK, 256, 0, stream>>>(x, norm1_g, norm1_b, xn);
    launch_gemm(xn, in_proj_w, in_proj_b, nullptr, nullptr, qkv,
                NTOK, 3 * E_, E_, 0, 0, stream);
    attention_kernel<<<dim3(S_, B_ * H_), 64, 0, stream>>>(qkv, ctx);
    launch_gemm(ctx, out_proj_w, out_proj_b, /*residual=*/x, nullptr, x1,
                NTOK, E_, E_, 0, 0, stream);

    // --- MoE block ---
    ln_kernel<<<NTOK, 256, 0, stream>>>(x1, norm2_g, norm2_b, xn);
    router_kernel<<<(NTOK + 255) / 256, 256, 0, stream>>>(
        xn, router_w, router_b, modality, coef, vm, tm);

    // 11 dense experts: 3 specialized (1.0 / vmask / tmask) + 8 general (coef).
    const float* LG[11]; const float* LB[11];
    const float* W1[11]; const float* B1[11];
    const float* W2[11]; const float* B2[11];
    const float* SCL[11];
    for (int e = 0; e < 3; ++e) {
        LG[e] = spec_ln_g + (size_t)e * E_;
        LB[e] = spec_ln_b + (size_t)e * E_;
        W1[e] = spec_w1 + (size_t)e * F_ * E_;
        B1[e] = spec_b1 + (size_t)e * F_;
        W2[e] = spec_w2 + (size_t)e * E_ * F_;
        B2[e] = spec_b2 + (size_t)e * E_;
    }
    SCL[0] = nullptr; SCL[1] = vm; SCL[2] = tm;
    for (int e = 0; e < NE_; ++e) {
        int i = 3 + e;
        LG[i] = gen_ln_g + (size_t)e * E_;
        LB[i] = gen_ln_b + (size_t)e * E_;
        W1[i] = gen_w1 + (size_t)e * F_ * E_;
        B1[i] = gen_b1 + (size_t)e * F_;
        W2[i] = gen_w2 + (size_t)e * E_ * F_;
        B2[i] = gen_b2 + (size_t)e * E_;
        SCL[i] = coef + (size_t)e * NTOK;
    }

    for (int i = 0; i < 11; ++i) {
        ln_kernel<<<NTOK, 256, 0, stream>>>(xn, LG[i], LB[i], xln);
        // h = GELU(xln @ w1^T + b1)
        launch_gemm(xln, W1[i], B1[i], nullptr, nullptr, h,
                    NTOK, F_, E_, /*gelu=*/1, /*accum=*/0, stream);
        // expert_out (+)= scale * (h @ w2^T + b2); first expert initializes eo.
        launch_gemm(h, W2[i], B2[i], nullptr, SCL[i], eo,
                    NTOK, E_, F_, /*gelu=*/0, /*accum=*/(i > 0) ? 1 : 0, stream);
    }

    // out = x1 + expert_out
    add_kernel<<<(NTOK * E_ + 255) / 256, 256, 0, stream>>>(x1, eo, out, NTOK * E_);
}
